// VDTBlock_40372692583163
// MI455X (gfx1250) — matrix operand order repaired and checked by reading, not compiled, not run on hardware
//
#include <hip/hip_runtime.h>
#include <hip/hip_bf16.h>
#include <math.h>

typedef __bf16 bf16;
typedef __attribute__((ext_vector_type(16))) __bf16 v16bf;
typedef __attribute__((ext_vector_type(8)))  float  v8f;
typedef int v4i __attribute__((vector_size(4 * sizeof(int))));   // matches builtin param

#define T_    16
#define H_    16
#define DH_   72
#define M_    1152
#define B_    2
#define N_    256
#define ROWS_ 8192           // B*T*N
#define ADAW_ 6912           // 6*M
#define SCALE_ 0.117851130198f  // 72^-0.5

#define BM 128
#define BN 128
#define BK 32

#if defined(__has_builtin)
#if __has_builtin(__builtin_amdgcn_global_load_async_to_lds_b128)
#define HAVE_ASYNC_LDS 1
#endif
#endif

union Frag { v16bf v; uint4 u[2]; };

// ---------------- elementwise f32 -> bf16 cast (weights) -------------------
__global__ __launch_bounds__(256) void k_cast_bf16(const float* __restrict__ s,
                                                   bf16* __restrict__ d, int n) {
  int i = blockIdx.x * 256 + threadIdx.x;
  if (i < n) d[i] = (bf16)s[i];
}

// ---------------- ada = silu(c4t) @ W_ada + b_ada --------------------------
__global__ __launch_bounds__(256) void k_ada(const float* __restrict__ c4t,
                                             const float* __restrict__ W,
                                             const float* __restrict__ bia,
                                             float* __restrict__ ada) {
  int i = blockIdx.x * 256 + threadIdx.x;
  if (i >= B_ * ADAW_) return;
  int b = i / ADAW_, col = i % ADAW_;
  float acc = bia[col];
  for (int k = 0; k < M_; ++k) {
    float v = c4t[b * M_ + k];
    float sv = v / (1.f + __expf(-v));
    acc += sv * W[(size_t)k * ADAW_ + col];
  }
  ada[i] = acc;
}

// ---------------- ek/ev projections: c @ {t_k,t_v,s_k,s_v} -----------------
// out layout: [which][B][T][M], which: 0=t_k 1=t_v 2=s_k 3=s_v
__global__ __launch_bounds__(256) void k_ekv(const float* __restrict__ c,
                                             const float* __restrict__ tk,
                                             const float* __restrict__ tv,
                                             const float* __restrict__ sk,
                                             const float* __restrict__ sv,
                                             float* __restrict__ out) {
  const int per = B_ * T_ * M_;
  int i = blockIdx.x * 256 + threadIdx.x;
  if (i >= 4 * per) return;
  int which = i / per, rem = i % per;
  int bt = rem / M_, col = rem % M_;
  const float* W = (which == 0) ? tk : (which == 1) ? tv : (which == 2) ? sk : sv;
  const float* cr = c + (size_t)bt * M_;
  float acc = 0.f;
  for (int k = 0; k < M_; ++k) acc += cr[k] * W[(size_t)k * M_ + col];
  out[i] = acc;
}

// ---------------- rotary in-place on temporal ek (pos = t) -----------------
__global__ __launch_bounds__(256) void k_rot_ek(float* __restrict__ e) {
  int i = blockIdx.x * 256 + threadIdx.x;          // B*T*M/2 pairs
  if (i >= B_ * T_ * M_ / 2) return;
  int col2 = i % (M_ / 2), bt = i / (M_ / 2);
  int t = bt % T_;
  int d = col2 % (DH_ / 2);
  float inv = __powf(10000.f, -2.f * d / (float)DH_);
  float ang = t * inv, cs = __cosf(ang), sn = __sinf(ang);
  float* p = e + (size_t)bt * M_ + col2 * 2;
  float a = p[0], b = p[1];
  p[0] = a * cs - b * sn;
  p[1] = b * cs + a * sn;
}

// ---------------- LayerNorm / modulate, fp32 -> bf16 -----------------------
// mode 0: gamma=gw[col], beta=gb[col];  mode 1: gamma=1+ada[b][scOff*M+col], beta=ada[b][shOff*M+col]
// transposeOut: remap (b,t,n) row -> (b,n,t) row
__global__ __launch_bounds__(128) void k_ln(const float* __restrict__ X,
                                            bf16* __restrict__ Y, float eps,
                                            const float* __restrict__ gw,
                                            const float* __restrict__ gb,
                                            const float* __restrict__ ada,
                                            int scOff, int shOff,
                                            int mode, int transposeOut) {
  int row = blockIdx.x;           // (b,t,n) order
  int tid = threadIdx.x;          // 128 threads, 9 elems each
  const float* xr = X + (size_t)row * M_;
  float v[9], s = 0.f, ss = 0.f;
#pragma unroll
  for (int j = 0; j < 9; ++j) {
    float x = xr[tid + j * 128];
    v[j] = x; s += x; ss += x * x;
  }
  __shared__ float r1[128], r2[128];
  r1[tid] = s; r2[tid] = ss;
  __syncthreads();
  for (int off = 64; off > 0; off >>= 1) {
    if (tid < off) { r1[tid] += r1[tid + off]; r2[tid] += r2[tid + off]; }
    __syncthreads();
  }
  float mu = r1[0] * (1.f / M_);
  float var = r2[0] * (1.f / M_) - mu * mu;
  float inv = rsqrtf(var + eps);
  int b = row >> 12;
  int orow = row;
  if (transposeOut) {
    int t = (row >> 8) & 15, n = row & 255;
    orow = (b << 12) + n * 16 + t;
  }
  bf16* yr = Y + (size_t)orow * M_;
#pragma unroll
  for (int j = 0; j < 9; ++j) {
    int col = tid + j * 128;
    float g, be;
    if (mode == 0) { g = gw[col]; be = gb[col]; }
    else { g = 1.f + ada[b * ADAW_ + scOff * M_ + col]; be = ada[b * ADAW_ + shOff * M_ + col]; }
    yr[col] = (bf16)(((v[j] - mu) * inv) * g + be);
  }
}

// ---------------- tiled bf16 WMMA GEMM, double-buffered, fused epilogues ---
// C(8192 x Ncols) = A(8192 x K, bf16) @ W(K x Ncols, bf16)
// Block tile 128x128x32, 8 waves each 32x64 (2x4 WMMA tiles, 8 wmma/step).
// A tile staged via async global->LDS DMA (ASYNCcnt) when available.
// emode 0: outB = bf16(acc)
// emode 1: outB = bf16(gelu_tanh(acc + bias[col]))
// emode 2: x4[remap(row)] = xin[remap(row)] + acc + bias[col]   (rows in (b,n,t) order)
// emode 3: x4[row] += gains[b][col] * acc
// emode 4: outF[row] = x4[row] + gains[b][col] * (acc + bias[col])
__global__ __launch_bounds__(256) void k_gemm(const bf16* __restrict__ A,
                                              const bf16* __restrict__ W,
                                              int K, int Ncols, int emode,
                                              bf16* __restrict__ outB,
                                              const float* __restrict__ bias,
                                              const float* __restrict__ xin,
                                              float* __restrict__ x4,
                                              const float* __restrict__ gains,
                                              float* __restrict__ outF) {
  __shared__ __align__(16) bf16 As[2][BM * BK];
  __shared__ __align__(16) bf16 Bs[2][BN * BK];
  int tid  = threadIdx.x;
  int bm0  = blockIdx.y * BM;
  int bn0  = blockIdx.x * BN;
  int lane = tid & 31, wave = tid >> 5;
  int wm = wave >> 1, wn = wave & 1;       // 4x2 waves, wave tile 32(M) x 64(N)
  int lr = lane & 15, lh = lane >> 4;

  int arow = tid >> 1, akoff = (tid & 1) * 16;   // A stage: 128 rows x 32 k, 16 elems/thread
  int brow = tid >> 3, bcol = (tid & 7) * 16;    // B stage: 32 k x 128 n, 16 elems/thread

  const bf16* gArow = A + (size_t)(bm0 + arow) * K + akoff;

  // ---- preload tile 0 (synchronous) ----
  {
    const uint4* ga = (const uint4*)gArow;
    uint4 a0 = ga[0], a1 = ga[1];
    *(uint4*)&As[0][arow * BK + akoff]     = a0;
    *(uint4*)&As[0][arow * BK + akoff + 8] = a1;
    const uint4* gb = (const uint4*)(W + (size_t)brow * Ncols + bn0 + bcol);
    uint4 b0 = gb[0], b1 = gb[1];
    const bf16* be0 = (const bf16*)&b0;
    const bf16* be1 = (const bf16*)&b1;
#pragma unroll
    for (int i = 0; i < 8; ++i) {
      Bs[0][(bcol + i) * BK + brow]     = be0[i];
      Bs[0][(bcol + 8 + i) * BK + brow] = be1[i];
    }
  }
  __syncthreads();

  v8f acc[2][4] = {};
  int nsteps = K / BK;
  for (int s = 0; s < nsteps; ++s) {
    int cur = s & 1, nxt = cur ^ 1;
    bool havenext = (s + 1) < nsteps;
    int k1 = (s + 1) * BK;
    uint4 pb0, pb1;
#ifndef HAVE_ASYNC_LDS
    uint4 pa0, pa1;
#endif
    if (havenext) {
#ifdef HAVE_ASYNC_LDS
      // async DMA global -> LDS for the A tile (overlaps with WMMA below)
      __builtin_amdgcn_global_load_async_to_lds_b128(
          (__attribute__((address_space(1))) v4i*)(void*)(gArow + k1),
          (__attribute__((address_space(3))) v4i*)(void*)&As[nxt][arow * BK + akoff],
          0, 0);
      __builtin_amdgcn_global_load_async_to_lds_b128(
          (__attribute__((address_space(1))) v4i*)(void*)(gArow + k1 + 8),
          (__attribute__((address_space(3))) v4i*)(void*)&As[nxt][arow * BK + akoff + 8],
          0, 0);
#else
      const uint4* ga = (const uint4*)(gArow + k1);
      pa0 = ga[0]; pa1 = ga[1];
#endif
      const uint4* gb = (const uint4*)(W + (size_t)(k1 + brow) * Ncols + bn0 + bcol);
      pb0 = gb[0]; pb1 = gb[1];
    }

    // fragment loads per CDNA5 16-bit WMMA VGPR layouts
    Frag af[2], bf4[4];
#pragma unroll
    for (int mt = 0; mt < 2; ++mt) {
      int row = wm * 32 + mt * 16 + lr;
      af[mt].u[0] = *(const uint4*)&As[cur][row * BK + lh * 8];       // K base .. +7
      af[mt].u[1] = *(const uint4*)&As[cur][row * BK + 16 + lh * 8];  // K base+16 .. +23
    }
#pragma unroll
    for (int nt = 0; nt < 4; ++nt) {
      int col = wn * 64 + nt * 16 + lr;
      bf4[nt].u[0] = *(const uint4*)&Bs[cur][col * BK + lh * 8];
      bf4[nt].u[1] = *(const uint4*)&Bs[cur][col * BK + 16 + lh * 8];
    }
#pragma unroll
    for (int mt = 0; mt < 2; ++mt)
#pragma unroll
      for (int nt = 0; nt < 4; ++nt)
        acc[mt][nt] = __builtin_amdgcn_wmma_f32_16x16x32_bf16(
            false, af[mt].v, false, bf4[nt].v, (short)0, acc[mt][nt], false, false);

    if (havenext) {
#ifndef HAVE_ASYNC_LDS
      *(uint4*)&As[nxt][arow * BK + akoff]     = pa0;
      *(uint4*)&As[nxt][arow * BK + akoff + 8] = pa1;
#endif
      const bf16* be0 = (const bf16*)&pb0;
      const bf16* be1 = (const bf16*)&pb1;
#pragma unroll
      for (int i = 0; i < 8; ++i) {
        Bs[nxt][(bcol + i) * BK + brow]     = be0[i];
        Bs[nxt][(bcol + 8 + i) * BK + brow] = be1[i];
      }
#ifdef HAVE_ASYNC_LDS
#if __has_builtin(__builtin_amdgcn_s_wait_asynccnt)
      __builtin_amdgcn_s_wait_asynccnt(0);
#else
      asm volatile("s_wait_asynccnt 0x0" ::: "memory");
#endif
#endif
      __syncthreads();
    }
  }

  // epilogue (C layout: VGPR r -> m = r + lh*8, n = lr within 16x16 tile)
#pragma unroll
  for (int mt = 0; mt < 2; ++mt)
#pragma unroll
    for (int nt = 0; nt < 4; ++nt) {
      int col = bn0 + wn * 64 + nt * 16 + lr;
#pragma unroll
      for (int r = 0; r < 8; ++r) {
        int row = bm0 + wm * 32 + mt * 16 + lh * 8 + r;
        float v = acc[mt][nt][r];
        if (emode == 0) {
          outB[(size_t)row * Ncols + col] = (bf16)v;
        } else if (emode == 1) {
          float h = v + bias[col];
          float u = h + 0.044715f * h * h * h;
          outB[(size_t)row * Ncols + col] =
              (bf16)(0.5f * h * (1.f + tanhf(0.7978845608f * u)));
        } else if (emode == 2) {
          int b = row >> 12, n = (row >> 4) & 255, t = row & 15;
          size_t xr = ((size_t)(b << 12) + t * 256 + n) * M_ + col;
          x4[xr] = xin[xr] + v + bias[col];
        } else if (emode == 3) {
          int b = row >> 12;
          size_t idx = (size_t)row * M_ + col;
          x4[idx] += gains[b * ADAW_ + col] * v;
        } else {
          int b = row >> 12;
          size_t idx = (size_t)row * M_ + col;
          outF[idx] = x4[idx] + gains[b * ADAW_ + col] * (v + bias[col]);
        }
      }
    }
}

// ---------------- temporal attention (online softmax, rotary) --------------
// qkv rows in (b,n,t) order, cols [q|k|v] each M_; ek/ev: [B][T][M] (ek rotary pre-applied)
__global__ __launch_bounds__(256, 1) void k_attn_t(const bf16* __restrict__ qkv,
                                                   const float* __restrict__ ek,
                                                   const float* __restrict__ ev,
                                                   bf16* __restrict__ out) {
  int bid = blockIdx.x;                 // b*256+n
  int b = bid >> 8, n = bid & 255;
  int tid = threadIdx.x;                // h*16 + t
  int h = tid >> 4, t = tid & 15;
  size_t qrow = (size_t)(b * N_ + n) * T_ + t;
  const bf16* qp = qkv + qrow * (3 * M_) + h * DH_;
  float q[DH_];
#pragma unroll
  for (int i = 0; i < DH_ / 2; ++i) {
    float inv = __powf(10000.f, -2.f * i / (float)DH_);
    float ang = t * inv, cs = __cosf(ang), sn = __sinf(ang);
    float a = (float)qp[2 * i] * SCALE_, c2 = (float)qp[2 * i + 1] * SCALE_;
    q[2 * i] = a * cs - c2 * sn;
    q[2 * i + 1] = c2 * cs + a * sn;
  }
  float m = -1e30f, l = 0.f, o[DH_];
#pragma unroll
  for (int d = 0; d < DH_; ++d) o[d] = 0.f;
  for (int j = 0; j < 2 * T_; ++j) {
    float s = 0.f;
    if (j < T_) {
      const float* kp = ek + ((size_t)b * T_ + j) * M_ + h * DH_;
      for (int d = 0; d < DH_; ++d) s += q[d] * kp[d];
    } else {
      int tj = j - T_;
      const bf16* kp = qkv + ((size_t)(b * N_ + n) * T_ + tj) * (3 * M_) + M_ + h * DH_;
      for (int i = 0; i < DH_ / 2; ++i) {
        float inv = __powf(10000.f, -2.f * i / (float)DH_);
        float ang = tj * inv, cs = __cosf(ang), sn = __sinf(ang);
        float a = (float)kp[2 * i], c2 = (float)kp[2 * i + 1];
        s += q[2 * i] * (a * cs - c2 * sn) + q[2 * i + 1] * (c2 * cs + a * sn);
      }
    }
    float mn = fmaxf(m, s);
    float f = __expf(m - mn), w = __expf(s - mn);
    l = l * f + w; m = mn;
    if (j < T_) {
      const float* vp = ev + ((size_t)b * T_ + j) * M_ + h * DH_;
      for (int d = 0; d < DH_; ++d) o[d] = o[d] * f + w * vp[d];
    } else {
      const bf16* vp = qkv + ((size_t)(b * N_ + n) * T_ + (j - T_)) * (3 * M_) + 2 * M_ + h * DH_;
      for (int d = 0; d < DH_; ++d) o[d] = o[d] * f + w * (float)vp[d];
    }
  }
  bf16* op = out + qrow * M_ + h * DH_;
  float li = 1.f / l;
  for (int d = 0; d < DH_; ++d) op[d] = (bf16)(o[d] * li);
}

// ---------------- spatial attention (online softmax, 257 keys) -------------
// qkv rows in (b,t,n) order; ek/ev: [B][T][M] single extra token (j==0)
__global__ __launch_bounds__(256, 1) void k_attn_s(const bf16* __restrict__ qkv,
                                                   const float* __restrict__ ek,
                                                   const float* __restrict__ ev,
                                                   bf16* __restrict__ out) {
  int bid = blockIdx.x;                 // b*256 + t*16 + h
  int b = bid >> 8, t = (bid >> 4) & 15, h = bid & 15;
  int n = threadIdx.x;
  size_t qrow = (size_t)(b * T_ + t) * N_ + n;
  const bf16* qp = qkv + qrow * (3 * M_) + h * DH_;
  float q[DH_];
#pragma unroll
  for (int d = 0; d < DH_; ++d) q[d] = (float)qp[d] * SCALE_;
  float m = -1e30f, l = 0.f, o[DH_];
#pragma unroll
  for (int d = 0; d < DH_; ++d) o[d] = 0.f;
  for (int j = 0; j < N_ + 1; ++j) {
    float s = 0.f;
    if (j == 0) {
      const float* kp = ek + ((size_t)b * T_ + t) * M_ + h * DH_;
      for (int d = 0; d < DH_; ++d) s += q[d] * kp[d];
    } else {
      const bf16* kp = qkv + ((size_t)(b * T_ + t) * N_ + (j - 1)) * (3 * M_) + M_ + h * DH_;
      for (int d = 0; d < DH_; ++d) s += q[d] * (float)kp[d];
    }
    float mn = fmaxf(m, s);
    float f = __expf(m - mn), w = __expf(s - mn);
    l = l * f + w; m = mn;
    if (j == 0) {
      const float* vp = ev + ((size_t)b * T_ + t) * M_ + h * DH_;
      for (int d = 0; d < DH_; ++d) o[d] = o[d] * f + w * vp[d];
    } else {
      const bf16* vp = qkv + ((size_t)(b * T_ + t) * N_ + (j - 1)) * (3 * M_) + 2 * M_ + h * DH_;
      for (int d = 0; d < DH_; ++d) o[d] = o[d] * f + w * (float)vp[d];
    }
  }
  bf16* op = out + qrow * M_ + h * DH_;
  float li = 1.f / l;
  for (int d = 0; d < DH_; ++d) op[d] = (bf16)(o[d] * li);
}

// ===========================================================================
extern "C" void kernel_launch(void* const* d_in, const int* in_sizes, int n_in,
                              void* d_out, int out_size, void* d_ws, size_t ws_size,
                              hipStream_t stream) {
  const float* x      = (const float*)d_in[0];
  const float* c4t    = (const float*)d_in[1];
  const float* c      = (const float*)d_in[2];
  const float* W_ada  = (const float*)d_in[3];
  const float* b_ada  = (const float*)d_in[4];
  const float* tn_w   = (const float*)d_in[5];
  const float* tn_b   = (const float*)d_in[6];
  const float* t_qkv  = (const float*)d_in[7];
  const float* t_k    = (const float*)d_in[8];
  const float* t_v    = (const float*)d_in[9];
  const float* t_out  = (const float*)d_in[10];
  const float* t_fc_w = (const float*)d_in[11];
  const float* t_fc_b = (const float*)d_in[12];
  const float* s_qkv  = (const float*)d_in[13];
  const float* s_k    = (const float*)d_in[14];
  const float* s_v    = (const float*)d_in[15];
  const float* s_out  = (const float*)d_in[16];
  const float* mlp_w1 = (const float*)d_in[17];
  const float* mlp_b1 = (const float*)d_in[18];
  const float* mlp_w2 = (const float*)d_in[19];
  const float* mlp_b2 = (const float*)d_in[20];
  float* dout = (float*)d_out;

  // workspace carve-up (~162 MB total)
  char* ws = (char*)d_ws;
  size_t off = 0;
  auto carve = [&](size_t bytes) { char* p = ws + off; off = (off + bytes + 255) & ~(size_t)255; return p; };
  float* x4   = (float*)carve((size_t)ROWS_ * M_ * 4);          // running residual
  bf16*  Abuf = (bf16*) carve((size_t)ROWS_ * M_ * 2);          // GEMM inputs (bf16)
  bf16*  Bbuf = (bf16*) carve((size_t)ROWS_ * 4 * M_ * 2);      // GEMM outputs (qkv / mlp h)
  bf16*  Cbuf = (bf16*) carve((size_t)ROWS_ * M_ * 2);          // attn outputs
  bf16*  wbuf = (bf16*) carve((size_t)M_ * 4 * M_ * 2);         // weight-bf16 scratch
  float* ada  = (float*)carve((size_t)B_ * ADAW_ * 4);
  float* ekv  = (float*)carve((size_t)4 * B_ * T_ * M_ * 4);    // [tk|tv|sk|sv]
  (void)ws_size; (void)in_sizes; (void)n_in; (void)out_size;

  const int per_ekv = B_ * T_ * M_;
  dim3 blk256(256), blk128(128);

  // conditioning
  k_ada<<<(B_ * ADAW_ + 255) / 256, blk256, 0, stream>>>(c4t, W_ada, b_ada, ada);
  k_ekv<<<(4 * per_ekv + 255) / 256, blk256, 0, stream>>>(c, t_k, t_v, s_k, s_v, ekv);
  k_rot_ek<<<(B_ * T_ * M_ / 2 + 255) / 256, blk256, 0, stream>>>(ekv);  // temporal ek

  // ---- temporal branch ----
  k_ln<<<ROWS_, blk128, 0, stream>>>(x, Abuf, 1e-5f, tn_w, tn_b, ada, 0, 0, 0, 1);
  k_cast_bf16<<<(M_ * 3 * M_ + 255) / 256, blk256, 0, stream>>>(t_qkv, wbuf, M_ * 3 * M_);
  k_gemm<<<dim3(27, 64), blk256, 0, stream>>>(Abuf, wbuf, M_, 3 * M_, 0, Bbuf,
                                              nullptr, nullptr, nullptr, nullptr, nullptr);
  k_attn_t<<<B_ * N_, blk256, 0, stream>>>(Bbuf, ekv, ekv + per_ekv, Cbuf);
  k_cast_bf16<<<(M_ * M_ + 255) / 256, blk256, 0, stream>>>(t_out, wbuf, M_ * M_);
  k_gemm<<<dim3(9, 64), blk256, 0, stream>>>(Cbuf, wbuf, M_, M_, 0, Abuf,
                                             nullptr, nullptr, nullptr, nullptr, nullptr);
  k_cast_bf16<<<(M_ * M_ + 255) / 256, blk256, 0, stream>>>(t_fc_w, wbuf, M_ * M_);
  k_gemm<<<dim3(9, 64), blk256, 0, stream>>>(Abuf, wbuf, M_, M_, 2, nullptr,
                                             t_fc_b, x, x4, nullptr, nullptr);

  // ---- spatial branch ----
  k_ln<<<ROWS_, blk128, 0, stream>>>(x4, Abuf, 1e-6f, nullptr, nullptr, ada, 1, 0, 1, 0);
  k_cast_bf16<<<(M_ * 3 * M_ + 255) / 256, blk256, 0, stream>>>(s_qkv, wbuf, M_ * 3 * M_);
  k_gemm<<<dim3(27, 64), blk256, 0, stream>>>(Abuf, wbuf, M_, 3 * M_, 0, Bbuf,
                                              nullptr, nullptr, nullptr, nullptr, nullptr);
  k_attn_s<<<B_ * T_ * H_, blk256, 0, stream>>>(Bbuf, ekv + 2 * per_ekv, ekv + 3 * per_ekv, Cbuf);
  k_cast_bf16<<<(M_ * M_ + 255) / 256, blk256, 0, stream>>>(s_out, wbuf, M_ * M_);
  k_gemm<<<dim3(9, 64), blk256, 0, stream>>>(Cbuf, wbuf, M_, M_, 3, nullptr,
                                             nullptr, nullptr, x4, ada + 2 * M_, nullptr);

  // ---- MLP ----
  k_ln<<<ROWS_, blk128, 0, stream>>>(x4, Abuf, 1e-6f, nullptr, nullptr, ada, 4, 3, 1, 0);
  k_cast_bf16<<<(M_ * 4 * M_ + 255) / 256, blk256, 0, stream>>>(mlp_w1, wbuf, M_ * 4 * M_);
  k_gemm<<<dim3(36, 64), blk256, 0, stream>>>(Abuf, wbuf, M_, 4 * M_, 1, Bbuf,
                                              mlp_b1, nullptr, nullptr, nullptr, nullptr);
  k_cast_bf16<<<(4 * M_ * M_ + 255) / 256, blk256, 0, stream>>>(mlp_w2, wbuf, 4 * M_ * M_);
  k_gemm<<<dim3(9, 64), blk256, 0, stream>>>(Bbuf, wbuf, 4 * M_, M_, 4, nullptr,
                                             mlp_b2, nullptr, x4, ada + 5 * M_, dout);
}
